// GenHash_7198365188181
// MI455X (gfx1250) — matrix operand order repaired and checked by reading, not compiled, not run on hardware
//
#include <hip/hip_runtime.h>

typedef __attribute__((ext_vector_type(16))) _Float16 half16;
typedef __attribute__((ext_vector_type(8)))  _Float16 half8;
typedef __attribute__((ext_vector_type(8)))  float    float8;

// ---------------------------------------------------------------------------
// helpers
// ---------------------------------------------------------------------------

// Branch-free tanh: 1 - 2/(exp(2x)+1), |x| clamped to 10 (tanh saturated).
// Lowers to v_exp_f32 + v_rcp_f32 + a few VALU ops; no EXEC divergence.
__device__ __forceinline__ float fast_tanh(float x) {
    float t = fminf(fmaxf(x, -10.0f), 10.0f);
    float e = __expf(2.0f * t);
    return 1.0f - 2.0f * __builtin_amdgcn_rcpf(e + 1.0f);
}

__device__ __forceinline__ float gelu_f(float x) {
    // jax.nn.gelu default (approximate=True, tanh form)
    const float k0 = 0.7978845608028654f;   // sqrt(2/pi)
    const float k1 = 0.044715f;
    float x3 = x * x * x;
    return 0.5f * x * (1.0f + fast_tanh(k0 * (x + k1 * x3)));
}

// Load a 16x32 f16 A-fragment (ISA 7.12.2): lane holds row (lane&15),
// K chunks [koff..koff+7] and [koff+16..koff+23] with koff=(lane>>4)*8.
// Caller passes p = row_base + k_chunk0; chunks are contiguous in memory.
__device__ __forceinline__ half16 load_frag_a(const _Float16* p) {
    half8 lo = *(const half8*)(p);
    half8 hi = *(const half8*)(p + 16);
    half16 r;
#pragma unroll
    for (int j = 0; j < 8; ++j) { r[j] = lo[j]; r[j + 8] = hi[j]; }
    return r;
}

__device__ __forceinline__ float8 wmma_f16(half16 a, half16 b, float8 c) {
    // D = A(16x32 f16) * B(32x16 f16) + C(16x16 f32)
    return __builtin_amdgcn_wmma_f32_16x16x32_f16(
        /*neg_a=*/false, a, /*neg_b=*/false, b,
        /*c_mod=*/(short)0, c, /*reuse_a=*/false, /*reuse_b=*/false);
}

// ---------------------------------------------------------------------------
// Pack a 128x128 (K x N) f32 weight into per-lane WMMA B-fragments (f16).
// Layout: [ntile(8)][kstep(4)][lane(32)][elem(16)], elem j <-> K = kstep*32 +
// (lane>>4)*8 + (j&7) + (j>>3)*16, N = ntile*16 + (lane&15).
// ---------------------------------------------------------------------------
__global__ __launch_bounds__(256) void k_pack128(const float* __restrict__ W,
                                                 _Float16* __restrict__ out) {
    int idx   = blockIdx.x * 256 + threadIdx.x;   // 16384 total
    int j     = idx & 15;
    int lane  = (idx >> 4) & 31;
    int ks    = (idx >> 9) & 3;
    int ntile = idx >> 11;
    int K = ks * 32 + (lane >> 4) * 8 + (j & 7) + (j >> 3) * 16;
    int n = ntile * 16 + (lane & 15);
    out[idx] = (_Float16)W[K * 128 + n];
}

// hW1: (64, 160, 256) -> [l][ntile(16)][kstep(5)][lane(32)][elem(16)]
__global__ __launch_bounds__(256) void k_pack_hw1(const float* __restrict__ W,
                                                  _Float16* __restrict__ out) {
    int idx  = blockIdx.x * 256 + threadIdx.x;    // 2,621,440 total
    int j    = idx & 15;
    int lane = (idx >> 4) & 31;
    int fk   = idx >> 9;          // (l*16 + ntile)*5 + ks
    int ks   = fk % 5;
    int tn   = fk / 5;
    int ntile = tn & 15;
    int l     = tn >> 4;
    int K = ks * 32 + (lane >> 4) * 8 + (j & 7) + (j >> 3) * 16;       // 0..159
    int n = ntile * 16 + (lane & 15);
    out[idx] = (_Float16)W[((size_t)l * 160 + K) * 256 + n];
}

// ---------------------------------------------------------------------------
// y-path: gather emb[y] -> y_embed (f32 output) ; y_h = gelu(ye@yW1+b)@yW2+b
// 128 rows per block, 8 waves, wave w owns M-tile w (rows 16w..16w+15).
// ---------------------------------------------------------------------------
__global__ __launch_bounds__(256) void k_ypath(
    const int* __restrict__ y, const float* __restrict__ emb,
    const _Float16* __restrict__ yW1p, const float* __restrict__ yb1,
    const _Float16* __restrict__ yW2p, const float* __restrict__ yb2,
    float* __restrict__ y_embed_out, _Float16* __restrict__ y_h_out) {
    __shared__ _Float16 Abuf[128 * 128];          // 32 KB, reused between GEMMs

    const int tid  = threadIdx.x;
    const int lane = tid & 31;
    const int w    = tid >> 5;
    const int b0   = blockIdx.x * 128;

    // gather: two threads per row, 64 contiguous floats each
    {
        int r     = tid >> 1;
        int half0 = (tid & 1) * 64;
        int b     = b0 + r;
        int label = y[b];
        const float* src = emb + label * 128 + half0;
        float* dst = y_embed_out + (size_t)b * 128 + half0;
#pragma unroll
        for (int j = 0; j < 64; j += 4) {
            float4 v = *(const float4*)(src + j);
            *(float4*)(dst + j) = v;
            Abuf[r * 128 + half0 + j + 0] = (_Float16)v.x;
            Abuf[r * 128 + half0 + j + 1] = (_Float16)v.y;
            Abuf[r * 128 + half0 + j + 2] = (_Float16)v.z;
            Abuf[r * 128 + half0 + j + 3] = (_Float16)v.w;
        }
    }
    __syncthreads();

    const int mrow = w * 16 + (lane & 15);
    const int koff = (lane >> 4) * 8;

    // GEMM1: A from LDS, B = packed yW1; gelu; write back into Abuf (f16)
    half16 a[4];
#pragma unroll
    for (int ks = 0; ks < 4; ++ks)
        a[ks] = load_frag_a(&Abuf[mrow * 128 + ks * 32 + koff]);

#pragma unroll
    for (int nt = 0; nt < 8; ++nt) {
        float8 acc = {};
#pragma unroll
        for (int ks = 0; ks < 4; ++ks) {
            half16 bf = *(const half16*)(yW1p + ((nt * 4 + ks) * 32 + lane) * 16);
            acc = wmma_f16(a[ks], bf, acc);
        }
        int col = nt * 16 + (lane & 15);
        float bias = yb1[col];
#pragma unroll
        for (int i = 0; i < 8; ++i) {
            int rr = w * 16 + i + (lane >> 4) * 8;
            Abuf[rr * 128 + col] = (_Float16)gelu_f(acc[i] + bias);
        }
    }
    __syncthreads();

    // GEMM2: y_h = mid @ yW2 + yb2 (no activation), f16 to workspace
#pragma unroll
    for (int ks = 0; ks < 4; ++ks)
        a[ks] = load_frag_a(&Abuf[mrow * 128 + ks * 32 + koff]);

#pragma unroll
    for (int nt = 0; nt < 8; ++nt) {
        float8 acc = {};
#pragma unroll
        for (int ks = 0; ks < 4; ++ks) {
            half16 bf = *(const half16*)(yW2p + ((nt * 4 + ks) * 32 + lane) * 16);
            acc = wmma_f16(a[ks], bf, acc);
        }
        int col = nt * 16 + (lane & 15);
        float bias = yb2[col];
#pragma unroll
        for (int i = 0; i < 8; ++i) {
            int rr = w * 16 + i + (lane >> 4) * 8;
            y_h_out[(size_t)(b0 + rr) * 128 + col] = (_Float16)(acc[i] + bias);
        }
    }
}

// ---------------------------------------------------------------------------
// main: per (l, 64-row b-tile):
//   H1 = gelu([y_h | e_h_l] @ hW1[l] + hb1[l])      (64x160 @ 160x256, WMMA)
//   out[b,l] = tanh(3*(H1 @ hW2[l] + hb2[l]))       (matvec via LDS+shuffle)
// 8 waves: wave w owns M-tile (w&3), N-tiles (w>>2)*8 .. +7.
// ---------------------------------------------------------------------------
__global__ __launch_bounds__(256) void k_main(
    const _Float16* __restrict__ y_h, const float* __restrict__ e,
    const float* __restrict__ eW1, const float* __restrict__ eb1,
    const float* __restrict__ eW2, const float* __restrict__ eb2,
    const _Float16* __restrict__ hW1p, const float* __restrict__ hb1,
    const float* __restrict__ hW2, const float* __restrict__ hb2,
    float* __restrict__ h_out) {
    __shared__ _Float16 H1[64 * 256];   // 32 KB
    __shared__ float sW2[256];
    __shared__ float sEW2[512];

    const int tid  = threadIdx.x;
    const int lane = tid & 31;
    const int w    = tid >> 5;
    const int l    = blockIdx.y;
    const int b0   = blockIdx.x * 64;

    sW2[tid]        = hW2[l * 256 + tid];
    sEW2[tid]       = eW2[tid];
    sEW2[tid + 256] = eW2[tid + 256];
    __syncthreads();

    const int mtile = w & 3;
    const int nb    = (w >> 2) * 8;
    const int mrow  = mtile * 16 + (lane & 15);
    const int b     = b0 + mrow;
    const int koff  = (lane >> 4) * 8;

    half16 a[5];
    // K-steps 0..3: rows of y_h, contiguous chunks straight from global
#pragma unroll
    for (int ks = 0; ks < 4; ++ks)
        a[ks] = load_frag_a(y_h + (size_t)b * 128 + ks * 32 + koff);

    // K-step 4: e_h[b, l, 0..31] computed on the fly (1->16 gelu ->32)
    {
        float x = e[b * 64 + l];
        float tg[16];
#pragma unroll
        for (int m = 0; m < 16; ++m) tg[m] = gelu_f(x * eW1[m] + eb1[m]);
#pragma unroll
        for (int j = 0; j < 8; ++j) {
            int k0 = koff + j, k1 = koff + 16 + j;
            float s0 = eb2[k0], s1 = eb2[k1];
#pragma unroll
            for (int m = 0; m < 16; ++m) {
                s0 += tg[m] * sEW2[m * 32 + k0];
                s1 += tg[m] * sEW2[m * 32 + k1];
            }
            a[4][j]     = (_Float16)s0;
            a[4][8 + j] = (_Float16)s1;
        }
    }

    const _Float16* bp = hW1p + (((size_t)l * 16 + nb) * 5) * 512;
#pragma unroll
    for (int nt = 0; nt < 8; ++nt) {
        float8 acc = {};
#pragma unroll
        for (int ks = 0; ks < 5; ++ks) {
            half16 bf = *(const half16*)(bp + (size_t)(nt * 5 + ks) * 512 + lane * 16);
            acc = wmma_f16(a[ks], bf, acc);
        }
        int col = (nb + nt) * 16 + (lane & 15);
        float bias = hb1[l * 256 + col];
#pragma unroll
        for (int i = 0; i < 8; ++i) {
            int rr = mtile * 16 + i + (lane >> 4) * 8;
            H1[rr * 256 + col] = (_Float16)gelu_f(acc[i] + bias);
        }
    }
    __syncthreads();

    // matvec: 4 threads per row, 64 cols each; reduce with lane shuffles
    {
        int row = tid >> 2;
        int q   = tid & 3;
        const _Float16* hp = &H1[row * 256 + q * 64];
        const float*    wp = &sW2[q * 64];
        float s = 0.0f;
#pragma unroll
        for (int c = 0; c < 64; ++c) s += (float)hp[c] * wp[c];
        s += __shfl_xor(s, 1, 32);
        s += __shfl_xor(s, 2, 32);
        if (q == 0) {
            h_out[(b0 + row) * 64 + l] = fast_tanh(3.0f * (s + hb2[l]));
        }
    }
}

// ---------------------------------------------------------------------------
// launch
// ---------------------------------------------------------------------------
extern "C" void kernel_launch(void* const* d_in, const int* in_sizes, int n_in,
                              void* d_out, int out_size, void* d_ws, size_t ws_size,
                              hipStream_t stream) {
    const int*   y   = (const int*)  d_in[0];
    const float* e   = (const float*)d_in[1];
    const float* emb = (const float*)d_in[2];
    const float* yW1 = (const float*)d_in[3];
    const float* yb1 = (const float*)d_in[4];
    const float* yW2 = (const float*)d_in[5];
    const float* yb2 = (const float*)d_in[6];
    const float* eW1 = (const float*)d_in[7];
    const float* eb1 = (const float*)d_in[8];
    const float* eW2 = (const float*)d_in[9];
    const float* eb2 = (const float*)d_in[10];
    const float* hW1 = (const float*)d_in[11];
    const float* hb1 = (const float*)d_in[12];
    const float* hW2 = (const float*)d_in[13];
    const float* hb2 = (const float*)d_in[14];

    float* h_out    = (float*)d_out;                 // (8192, 64)
    float* yemb_out = (float*)d_out + 8192 * 64;     // (8192, 128)

    char* ws = (char*)d_ws;
    _Float16* yW1p = (_Float16*)(ws);                                  // 32 KB
    _Float16* yW2p = (_Float16*)(ws + (32 << 10));                     // 32 KB
    _Float16* hW1p = (_Float16*)(ws + (64 << 10));                     // 5 MB
    _Float16* y_h  = (_Float16*)(ws + (64 << 10) + (size_t)5242880);   // 2 MB

    k_pack128 <<<   64, 256, 0, stream>>>(yW1, yW1p);
    k_pack128 <<<   64, 256, 0, stream>>>(yW2, yW2p);
    k_pack_hw1<<<10240, 256, 0, stream>>>(hW1, hW1p);
    k_ypath   <<<   64, 256, 0, stream>>>(y, emb, yW1p, yb1, yW2p, yb2,
                                          yemb_out, y_h);
    k_main    <<<dim3(128, 64), 256, 0, stream>>>(y_h, e, eW1, eb1, eW2, eb2,
                                                  hW1p, hb1, hW2, hb2, h_out);
}
